// Attention_12369505812708
// MI455X (gfx1250) — compile-verified, matrix operand order checked by reading, not run on hardware
//
#include <hip/hip_runtime.h>

// ---------------- problem constants ----------------
constexpr int Bq   = 2;
constexpr int Lq   = 2048;
constexpr int Dq   = 2048;
constexpr int Hq   = 16;
constexpr int KVHq = 4;
constexpr int HDq  = 128;          // head dim
constexpr int Mq   = Bq * Lq;      // 4096 rows
constexpr int KVDq = KVHq * HDq;   // 512

typedef __attribute__((ext_vector_type(16))) __bf16 v16bf;
typedef __attribute__((ext_vector_type(8)))  float  v8f;
typedef __attribute__((ext_vector_type(8)))  unsigned short ush8;

__device__ inline unsigned short f2bf(float f) {
  union { float f; unsigned int u; } x; x.f = f;
  unsigned int u = x.u;
  unsigned int r = (u + 0x7FFFu + ((u >> 16) & 1u)) >> 16;   // RNE
  return (unsigned short)r;
}

__device__ inline v8f zero8() {
  v8f z;
#pragma unroll
  for (int i = 0; i < 8; ++i) z[i] = 0.0f;
  return z;
}

// Load a 16-bit A/B WMMA fragment (16x32 / 32x16) from K-contiguous storage.
// base points at the lane's row (32 halves); kb = 0 or 8 (per half-wave).
__device__ inline v16bf ld_frag(const unsigned short* base, int kb) {
  union { ush8 h[2]; v16bf v; } f;
  f.h[0] = *(const ush8*)(base + kb);        // K = kb .. kb+7
  f.h[1] = *(const ush8*)(base + kb + 16);   // K = kb+16 .. kb+23
  return f.v;
}

__device__ inline v8f wmma_bf16(v16bf a, v16bf b, v8f c) {
  return __builtin_amdgcn_wmma_f32_16x16x32_bf16(
      /*neg_a=*/false, a, /*neg_b=*/false, b,
      /*c_mod=*/(short)0, c, /*reuse_a=*/false, /*reuse_b=*/false);
}

// ---------------- elementwise prep kernels ----------------
__global__ void cast_f32_bf16(const float* __restrict__ in,
                              unsigned short* __restrict__ out, size_t n) {
  size_t i = (size_t)blockIdx.x * blockDim.x + threadIdx.x;
  if (i < n) out[i] = f2bf(in[i]);
}

// W [K][N] row-major fp32  ->  Wt [N][K] bf16 (column-major for B-fragments)
__global__ void transpose_cast_w(const float* __restrict__ W,
                                 unsigned short* __restrict__ Wt,
                                 int Kd, int Nd) {
  size_t i = (size_t)blockIdx.x * blockDim.x + threadIdx.x;
  size_t total = (size_t)Kd * Nd;
  if (i >= total) return;
  int n = (int)(i % Nd);
  int k = (int)(i / Nd);
  Wt[(size_t)n * Kd + k] = f2bf(W[i]);
}

// RoPE over pairs + cast to bf16 (in/out layout [B][L][Hh][HD]); scale folded in.
__global__ void rope_cast(const float* __restrict__ in,
                          const float* __restrict__ cos_t,
                          const float* __restrict__ sin_t,
                          unsigned short* __restrict__ out,
                          int Hh, float scale) {
  size_t i = (size_t)blockIdx.x * blockDim.x + threadIdx.x;
  size_t total = (size_t)Bq * Lq * Hh * (HDq / 2);
  if (i >= total) return;
  int r = (int)(i % (HDq / 2));
  size_t rest = i / (HDq / 2);
  int hh = (int)(rest % Hh);
  size_t bl = rest / Hh;              // b*L + l
  int l = (int)(bl % Lq);
  size_t base = (bl * Hh + hh) * HDq;
  float tr = in[base + 2 * r], ti = in[base + 2 * r + 1];
  float c = cos_t[(size_t)l * (HDq / 2) + r];
  float s = sin_t[(size_t)l * (HDq / 2) + r];
  out[base + 2 * r]     = f2bf((tr * c - ti * s) * scale);
  out[base + 2 * r + 1] = f2bf((tr * s + ti * c) * scale);
}

// V [B][L][KVH][HD] fp32 -> Vt [B][KVH][HD][L] bf16 (so P*V B-frags are contiguous)
__global__ void vcast_transpose(const float* __restrict__ in,
                                unsigned short* __restrict__ out) {
  size_t i = (size_t)blockIdx.x * blockDim.x + threadIdx.x;
  size_t total = (size_t)Bq * Lq * KVHq * HDq;
  if (i >= total) return;
  int hd = (int)(i % HDq);
  size_t rest = i / HDq;
  int kv = (int)(rest % KVHq);
  size_t bl = rest / KVHq;            // b*L + l
  int l = (int)(bl % Lq);
  int b = (int)(bl / Lq);
  out[(((size_t)b * KVHq + kv) * HDq + hd) * Lq + l] = f2bf(in[i]);
}

// ---------------- bf16 WMMA GEMM: C[M][N] = A[M][K] * Bt[N][K]^T + bias ----------------
// Block tile 128x128, 8 waves, wave tile 32x64, K-step 32, double-buffered LDS.
constexpr int TBM = 128, TBN = 128, TBK = 32;

__global__ __launch_bounds__(256)
void gemm_bf16(const unsigned short* __restrict__ A,   // [M][K] bf16
               const unsigned short* __restrict__ Bt,  // [N][K] bf16
               const float* __restrict__ bias,         // [N]
               float* __restrict__ C,                  // [M][N] fp32
               int M, int N, int K) {
  __shared__ __align__(16) unsigned short sA[2][TBM * TBK];  // 2 x 8 KB
  __shared__ __align__(16) unsigned short sB[2][TBN * TBK];  // 2 x 8 KB
  const int tid  = threadIdx.x;
  const int wave = tid >> 5, lane = tid & 31;
  const int wm = wave & 3;        // 0..3 -> M offset
  const int wn = wave >> 2;       // 0..1 -> N offset (64 wide)
  const int m0 = blockIdx.y * TBM;
  const int n0 = blockIdx.x * TBN;
  const int cl = lane & 15;
  const int lh = lane >> 4;
  const int kb = lh ? 8 : 0;

  // staging chunk coordinates (each thread moves 2 x 16B for A and for B)
  const int ra0 = tid >> 2,          oa0 = (tid & 3) * 8;
  const int ra1 = (tid + 256) >> 2,  oa1 = oa0;

  v8f acc[2][4];
#pragma unroll
  for (int mi = 0; mi < 2; ++mi)
#pragma unroll
    for (int ni = 0; ni < 4; ++ni) acc[mi][ni] = zero8();

  const int nK = K / TBK;

  // preload tile 0
  uint4 pa0 = *(const uint4*)&A [(size_t)(m0 + ra0) * K + oa0];
  uint4 pa1 = *(const uint4*)&A [(size_t)(m0 + ra1) * K + oa1];
  uint4 pb0 = *(const uint4*)&Bt[(size_t)(n0 + ra0) * K + oa0];
  uint4 pb1 = *(const uint4*)&Bt[(size_t)(n0 + ra1) * K + oa1];
  *(uint4*)&sA[0][ra0 * TBK + oa0] = pa0;
  *(uint4*)&sA[0][ra1 * TBK + oa1] = pa1;
  *(uint4*)&sB[0][ra0 * TBK + oa0] = pb0;
  *(uint4*)&sB[0][ra1 * TBK + oa1] = pb1;
  __syncthreads();

  for (int kt = 0; kt < nK; ++kt) {
    const int cur = kt & 1, nxt = cur ^ 1;
    const bool more = (kt + 1) < nK;
    if (more) {                           // issue next tile's global loads early
      const int k0 = (kt + 1) * TBK;
      pa0 = *(const uint4*)&A [(size_t)(m0 + ra0) * K + k0 + oa0];
      pa1 = *(const uint4*)&A [(size_t)(m0 + ra1) * K + k0 + oa1];
      pb0 = *(const uint4*)&Bt[(size_t)(n0 + ra0) * K + k0 + oa0];
      pb1 = *(const uint4*)&Bt[(size_t)(n0 + ra1) * K + k0 + oa1];
    }

    // fragments for this tile
    v16bf a[2], bfr[4];
#pragma unroll
    for (int mi = 0; mi < 2; ++mi)
      a[mi] = ld_frag(&sA[cur][(wm * 32 + mi * 16 + cl) * TBK], kb);
#pragma unroll
    for (int ni = 0; ni < 4; ++ni)
      bfr[ni] = ld_frag(&sB[cur][(wn * 64 + ni * 16 + cl) * TBK], kb);

#pragma unroll
    for (int mi = 0; mi < 2; ++mi)
#pragma unroll
      for (int ni = 0; ni < 4; ++ni)
        acc[mi][ni] = wmma_bf16(a[mi], bfr[ni], acc[mi][ni]);

    if (more) {                           // land next tile in the other buffer
      *(uint4*)&sA[nxt][ra0 * TBK + oa0] = pa0;
      *(uint4*)&sA[nxt][ra1 * TBK + oa1] = pa1;
      *(uint4*)&sB[nxt][ra0 * TBK + oa0] = pb0;
      *(uint4*)&sB[nxt][ra1 * TBK + oa1] = pb1;
    }
    __syncthreads();                      // one barrier per K-step
  }

  // epilogue: C-layout store + bias
#pragma unroll
  for (int ni = 0; ni < 4; ++ni) {
    const int col = n0 + wn * 64 + ni * 16 + cl;
    const float bz = bias[col];
#pragma unroll
    for (int mi = 0; mi < 2; ++mi) {
#pragma unroll
      for (int vi = 0; vi < 8; ++vi) {
        const int row = m0 + wm * 32 + mi * 16 + vi + 8 * lh;
        C[(size_t)row * N + col] = acc[mi][ni][vi] + bz;
      }
    }
  }
}

// ---------------- causal GQA flash attention (bf16 WMMA, fp32 online softmax) ----------------
// q [B][L][H][HD] bf16 (pre-scaled), k [B][L][KVH][HD] bf16, vt [B][KVH][HD][L] bf16
// out [B][L][H*HD] bf16
__global__ __launch_bounds__(128)
void attn_flash(const unsigned short* __restrict__ qb,
                const unsigned short* __restrict__ kbuf,
                const unsigned short* __restrict__ vtb,
                unsigned short* __restrict__ ob) {
  __shared__ __align__(16) unsigned short sK[32 * HDq];     // [key][hd]  8 KB
  __shared__ __align__(16) unsigned short sV[HDq * 32];     // [hd][key]  8 KB
  __shared__ __align__(16) unsigned short sP[4][16 * 32];   // per-wave P  4 KB

  const int b = blockIdx.z, h = blockIdx.y, qblk = blockIdx.x;
  const int kvh = h >> 2;                       // N_REP = 4
  const int tid = threadIdx.x, wave = tid >> 5, lane = tid & 31;
  const int cl = lane & 15, lh = lane >> 4;
  const int kb = lh ? 8 : 0;
  const int q0 = qblk * 64 + wave * 16;
  const int qrow = q0 + cl;                     // A-frag row for this lane

  // Q fragments: 4 K-chunks of 32 over HD=128
  v16bf aq[4];
  const unsigned short* qptr = &qb[(((size_t)b * Lq + qrow) * Hq + h) * HDq];
#pragma unroll
  for (int c = 0; c < 4; ++c) aq[c] = ld_frag(qptr + c * 32, kb);

  v8f oacc[8];
  float m_run[8], l_run[8];
#pragma unroll
  for (int i = 0; i < 8; ++i) { oacc[i] = zero8(); m_run[i] = -3.0e38f; l_run[i] = 0.0f; }

  const int nkb = qblk * 2 + 2;                 // causal: keys <= q0+63

  for (int t = 0; t < nkb; ++t) {
    const int key0 = t * 32;
#pragma unroll
    for (int i = 0; i < 4; ++i) {               // stage K tile: 512 x 16B
      int ch = tid + i * 128;
      int r = ch >> 4, o = (ch & 15) * 8;
      *(uint4*)&sK[r * HDq + o] =
          *(const uint4*)&kbuf[(((size_t)b * Lq + key0 + r) * KVHq + kvh) * HDq + o];
    }
#pragma unroll
    for (int i = 0; i < 4; ++i) {               // stage Vt tile: 512 x 16B
      int ch = tid + i * 128;
      int r = ch >> 2, o = (ch & 3) * 8;
      *(uint4*)&sV[r * 32 + o] =
          *(const uint4*)&vtb[(((size_t)b * KVHq + kvh) * HDq + r) * Lq + key0 + o];
    }
    __syncthreads();

    // S = Q * K^T  (two 16x16 sub-tiles over 32 keys)
    v8f s[2];
#pragma unroll
    for (int n = 0; n < 2; ++n) {
      s[n] = zero8();
#pragma unroll
      for (int c = 0; c < 4; ++c) {
        v16bf bf = ld_frag(&sK[(n * 16 + cl) * HDq + c * 32], kb);
        s[n] = wmma_bf16(aq[c], bf, s[n]);
      }
    }

    // online softmax update (per C-layout row vi; reduce across 16 column lanes)
#pragma unroll
    for (int vi = 0; vi < 8; ++vi) {
      const int rg = q0 + vi + 8 * lh;          // global query row
      float v0 = (key0 + cl      <= rg) ? s[0][vi] : -3.0e38f;
      float v1 = (key0 + 16 + cl <= rg) ? s[1][vi] : -3.0e38f;
      float t0 = fmaxf(v0, v1);
      t0 = fmaxf(t0, __shfl_xor(t0, 1));
      t0 = fmaxf(t0, __shfl_xor(t0, 2));
      t0 = fmaxf(t0, __shfl_xor(t0, 4));
      t0 = fmaxf(t0, __shfl_xor(t0, 8));
      float mnew  = fmaxf(m_run[vi], t0);
      float alpha = 1.0f, p0 = 0.0f, p1 = 0.0f;
      if (mnew > -1.0e38f) {
        alpha = __expf(m_run[vi] - mnew);
        p0 = __expf(v0 - mnew);
        p1 = __expf(v1 - mnew);
      }
      float rs = p0 + p1;
      rs += __shfl_xor(rs, 1);
      rs += __shfl_xor(rs, 2);
      rs += __shfl_xor(rs, 4);
      rs += __shfl_xor(rs, 8);
      l_run[vi] = l_run[vi] * alpha + rs;
      m_run[vi] = mnew;
#pragma unroll
      for (int nt = 0; nt < 8; ++nt) oacc[nt][vi] *= alpha;
      const int prow = vi + 8 * lh;             // C-layout -> row-major P in LDS
      sP[wave][prow * 32 + cl]      = f2bf(p0);
      sP[wave][prow * 32 + 16 + cl] = f2bf(p1);
    }

    // O += P * V   (per-wave LDS round-trip gives the A-matrix layout; LDS is in-order per wave)
    v16bf ap = ld_frag(&sP[wave][cl * 32], kb);
#pragma unroll
    for (int nt = 0; nt < 8; ++nt) {
      v16bf bf = ld_frag(&sV[(nt * 16 + cl) * 32], kb);
      oacc[nt] = wmma_bf16(ap, bf, oacc[nt]);
    }
    __syncthreads();
  }

  // normalize and emit bf16 in [B][L][H*HD] layout for the O-projection GEMM
#pragma unroll
  for (int vi = 0; vi < 8; ++vi) {
    const int rg = q0 + vi + 8 * lh;
    const float inv = (l_run[vi] > 0.0f) ? 1.0f / l_run[vi] : 0.0f;
#pragma unroll
    for (int nt = 0; nt < 8; ++nt) {
      ob[(((size_t)b * Lq + rg) * Hq + h) * HDq + nt * 16 + cl] =
          f2bf(oacc[nt][vi] * inv);
    }
  }
}

// ---------------- host-side launch ----------------
extern "C" void kernel_launch(void* const* d_in, const int* in_sizes, int n_in,
                              void* d_out, int out_size, void* d_ws, size_t ws_size,
                              hipStream_t stream) {
  const float* x       = (const float*)d_in[0];
  const float* f_cos   = (const float*)d_in[1];
  const float* f_sin   = (const float*)d_in[2];
  const float* Wq      = (const float*)d_in[3];
  const float* bq      = (const float*)d_in[4];
  const float* Wk      = (const float*)d_in[5];
  const float* bk      = (const float*)d_in[6];
  const float* Wv      = (const float*)d_in[7];
  const float* bv      = (const float*)d_in[8];
  const float* Wo      = (const float*)d_in[9];
  const float* bo      = (const float*)d_in[10];
  float* out = (float*)d_out;

  // workspace carve-up (bytes, 256-aligned sizes)
  char* w = (char*)d_ws;
  auto take = [&](size_t bytes) { char* p = w; w += (bytes + 255) & ~(size_t)255; return p; };
  unsigned short* x_bf   = (unsigned short*)take((size_t)Mq * Dq * 2);
  unsigned short* WqT    = (unsigned short*)take((size_t)Dq * Dq * 2);
  unsigned short* WkT    = (unsigned short*)take((size_t)KVDq * Dq * 2);
  unsigned short* WvT    = (unsigned short*)take((size_t)KVDq * Dq * 2);
  unsigned short* WoT    = (unsigned short*)take((size_t)Dq * Dq * 2);
  float*          q_f    = (float*)take((size_t)Mq * Dq * 4);
  float*          k_f    = (float*)take((size_t)Mq * KVDq * 4);
  float*          v_f    = (float*)take((size_t)Mq * KVDq * 4);
  unsigned short* q_bf   = (unsigned short*)take((size_t)Mq * Dq * 2);
  unsigned short* k_bf   = (unsigned short*)take((size_t)Mq * KVDq * 2);
  unsigned short* vt_bf  = (unsigned short*)take((size_t)Mq * KVDq * 2);
  unsigned short* att_bf = (unsigned short*)take((size_t)Mq * Dq * 2);

  const int TPB = 256;
  auto blocks = [&](size_t n) { return (unsigned)((n + TPB - 1) / TPB); };

  // 1) casts / transposes to bf16
  cast_f32_bf16<<<blocks((size_t)Mq * Dq), TPB, 0, stream>>>(x, x_bf, (size_t)Mq * Dq);
  transpose_cast_w<<<blocks((size_t)Dq * Dq), TPB, 0, stream>>>(Wq, WqT, Dq, Dq);
  transpose_cast_w<<<blocks((size_t)Dq * KVDq), TPB, 0, stream>>>(Wk, WkT, Dq, KVDq);
  transpose_cast_w<<<blocks((size_t)Dq * KVDq), TPB, 0, stream>>>(Wv, WvT, Dq, KVDq);
  transpose_cast_w<<<blocks((size_t)Dq * Dq), TPB, 0, stream>>>(Wo, WoT, Dq, Dq);

  // 2) projections (WMMA GEMMs): block tile 128x128
  gemm_bf16<<<dim3(Dq / TBN, Mq / TBM), TPB, 0, stream>>>(x_bf, WqT, bq, q_f, Mq, Dq, Dq);
  gemm_bf16<<<dim3(KVDq / TBN, Mq / TBM), TPB, 0, stream>>>(x_bf, WkT, bk, k_f, Mq, KVDq, Dq);
  gemm_bf16<<<dim3(KVDq / TBN, Mq / TBM), TPB, 0, stream>>>(x_bf, WvT, bv, v_f, Mq, KVDq, Dq);

  // 3) RoPE + cast (Q carries the 1/sqrt(HD) softmax scale), V transpose-cast
  const float qscale = 0.08838834764831845f;  // 1/sqrt(128)
  rope_cast<<<blocks((size_t)Mq * Hq * (HDq / 2)), TPB, 0, stream>>>(q_f, f_cos, f_sin, q_bf, Hq, qscale);
  rope_cast<<<blocks((size_t)Mq * KVHq * (HDq / 2)), TPB, 0, stream>>>(k_f, f_cos, f_sin, k_bf, KVHq, 1.0f);
  vcast_transpose<<<blocks((size_t)Mq * KVDq), TPB, 0, stream>>>(v_f, vt_bf);

  // 4) causal GQA flash attention
  attn_flash<<<dim3(Lq / 64, Hq, Bq), 128, 0, stream>>>(q_bf, k_bf, vt_bf, att_bf);

  // 5) output projection -> fp32 d_out
  gemm_bf16<<<dim3(Dq / TBN, Mq / TBM), TPB, 0, stream>>>(att_bf, WoT, bo, out, Mq, Dq, Dq);

  (void)in_sizes; (void)n_in; (void)out_size; (void)ws_size;
}
